// FasterRCNNModel_53815940219179
// MI455X (gfx1250) — matrix operand order repaired and checked by reading, not compiled, not run on hardware
//
#include <hip/hip_runtime.h>
#include <hip/hip_bf16.h>

// ---------------------------------------------------------------------------
// Faster-RCNN-lite forward for MI455X (gfx1250, wave32, WMMA).
//  * convs 2-4 + FC2: f16 WMMA (v_wmma_f32_16x16x32_f16), f32 accumulate
//  * activations kept in f16 end-to-end (HBM traffic ~halved)
//  * conv staging: one full tap (16px x Cin) per LDS barrier pair; staged via
//    GLOBAL_LOAD_ASYNC_TO_LDS_B128 (ASYNCcnt) when the builtin is available
//  * fc1 branch of the reference is dead (never returned) -> skipped
// ---------------------------------------------------------------------------

typedef __attribute__((ext_vector_type(16))) _Float16 v16h;
typedef __attribute__((ext_vector_type(8)))  _Float16 v8h;
typedef __attribute__((ext_vector_type(4)))  _Float16 v4h;
typedef __attribute__((ext_vector_type(8)))  float    v8f;
typedef __attribute__((ext_vector_type(4)))  int      v4i;

#define DEV __device__ __forceinline__

#if defined(__gfx1250__) && \
    __has_builtin(__builtin_amdgcn_global_load_async_to_lds_b128) && \
    __has_builtin(__builtin_amdgcn_s_wait_asynccnt)
#define HAVE_ASYNC_LDS 1
// builtin signature: void(v4i addrspace(1)*, v4i addrspace(3)*, imm int, imm int)
typedef __attribute__((address_space(1))) v4i* g128_ptr;
typedef __attribute__((address_space(3))) v4i* l128_ptr;
// 16-byte global -> LDS async copy (per-lane; tracked by ASYNCcnt)
DEV void async_copy16(const void* g, void* l) {
  __builtin_amdgcn_global_load_async_to_lds_b128(
      (g128_ptr)(uintptr_t)g, (l128_ptr)(uint32_t)(uintptr_t)l, 0, 0);
}
#else
#define HAVE_ASYNC_LDS 0
#endif

// ----- WMMA fragment loaders (contiguous k order, same for A and B) ----------------
// A (16x32 f16, row-major [16][ldk]): lanes 0-15 -> row m, K 0..15;
// lanes 16-31 -> row m, K 16..31.
DEV v16h load_fragA_k32(const _Float16* __restrict__ row_base, int ldk, int k0) {
  const int lane = threadIdx.x & 31;
  const int m  = lane & 15;
  const int kh = (lane < 16) ? 0 : 16;
  const _Float16* p = row_base + (size_t)m * ldk + k0 + kh;
  return *(const v16h*)(p);
}

// B (32x16 f16, stored K-contiguous per column: [N][ldk]):
// lanes 0-15 -> col n, K 0..15; lanes 16-31 -> col n, K 16..31.
DEV v16h load_fragB_k32(const _Float16* __restrict__ col_base, int ldk, int k0) {
  const int lane = threadIdx.x & 31;
  const int n  = lane & 15;
  const int kb = (lane < 16) ? 0 : 16;
  return *(const v16h*)(col_base + (size_t)n * ldk + k0 + kb);
}

// ----- weight packing -------------------------------------------------------
// conv HWIO (3,3,Cin,Cout) f32 -> [tap][Cout][Cin] f16 (B-fragment friendly)
__global__ __launch_bounds__(256)
void pack_conv_w(const float* __restrict__ w, _Float16* __restrict__ wp,
                 int Cin, int Cout) {
  int i = blockIdx.x * 256 + threadIdx.x;
  int total = 9 * Cin * Cout;
  if (i >= total) return;
  int ci = i % Cin;
  int j  = i / Cin;
  int co = j % Cout;
  int t  = j / Cout;
  wp[i] = (_Float16)w[((size_t)t * Cin + ci) * Cout + co];
}

// fc (K,N) f32 -> [N][K] f16
__global__ __launch_bounds__(256)
void pack_fc_w(const float* __restrict__ w, _Float16* __restrict__ wp,
               int K, int N) {
  int i = blockIdx.x * 256 + threadIdx.x;
  if (i >= K * N) return;
  int k = i % K;
  int n = i / K;
  wp[i] = (_Float16)w[(size_t)k * N + n];
}

// ----- conv1: Cin=3, stride 2, 512->256, direct (K=27 too small for WMMA) --
__global__ __launch_bounds__(256)
void conv1_direct_kernel(const float* __restrict__ x, const float* __restrict__ w,
                         const float* __restrict__ bias, _Float16* __restrict__ out,
                         int total) {
  int idx = blockIdx.x * 256 + threadIdx.x;
  if (idx >= total) return;
  const int co = idx & 63;
  int pix = idx >> 6;
  const int ox = pix & 255; pix >>= 8;
  const int oy = pix & 255; pix >>= 8;
  const int bb = pix;
  float acc = bias[co];
#pragma unroll
  for (int ky = 0; ky < 3; ++ky) {
    int iy = oy * 2 + ky;
    if (iy >= 512) continue;
#pragma unroll
    for (int kx = 0; kx < 3; ++kx) {
      int ix = ox * 2 + kx;
      if (ix >= 512) continue;
      const float* xp = x + (((size_t)bb * 512 + iy) * 512 + ix) * 3;
      const float* wp = w + ((size_t)(ky * 3 + kx) * 3) * 64 + co;
      acc += xp[0] * wp[0] + xp[1] * wp[64] + xp[2] * wp[128];
    }
  }
  acc = acc > 0.f ? acc : 0.f;
  out[idx] = (_Float16)acc;
}

// ----- convs 2-4: implicit-GEMM WMMA, stride 2, 3x3, SAME (pad_lo=0,hi=1) --
// Block = 128 threads = 4 waves; tile = 16 output pixels x 64 Cout.
// Per tap: stage 16 x CIN f16 activations into LDS once (async when
// available), then issue CIN/32 WMMAs per wave between a single barrier pair.
template <int CIN, int COUT>
__global__ __launch_bounds__(128)
void conv3x3_s2_wmma_kernel(const _Float16* __restrict__ in,   // [B][Hi][Wi][CIN] f16
                            const _Float16* __restrict__ wpk,  // [9][COUT][CIN]  f16
                            const float* __restrict__ bias,
                            _Float16* __restrict__ out,        // [B][Ho][Wo][COUT] f16
                            int Hi, int Wi, int Ho, int Wo) {
  __shared__ _Float16 sA[16 * CIN];
  const int tid  = threadIdx.x;
  const int wave = tid >> 5;
  const int lane = tid & 31;
  const int pix0 = blockIdx.x * 16;
  const int co0  = blockIdx.y * 64 + wave * 16;

  // cooperative loader role: 16 pixels x 8 column-groups (16B units)
  const int p = tid >> 3;
  const int g = tid & 7;
  const int pix = pix0 + p;
  const int hw  = Ho * Wo;
  const int b   = pix / hw;
  const int rem = pix - b * hw;
  const int oy  = rem / Wo;
  const int ox  = rem - oy * Wo;

  v8f acc = {};
  for (int t = 0; t < 9; ++t) {
    const int ky = t / 3, kx = t - ky * 3;
    const int iy = oy * 2 + ky;            // SAME w/ stride2: pad_lo = 0
    const int ix = ox * 2 + kx;
    const bool inb = (iy < Hi) && (ix < Wi);
    const size_t ibase = (((size_t)b * Hi + iy) * Wi + ix) * CIN;
    const _Float16* wbase = wpk + ((size_t)t * COUT + co0) * CIN;
    if (t < 8)  // hint next tap's weights toward L2 (global_prefetch_b8)
      __builtin_prefetch(wpk + ((size_t)(t + 1) * COUT + co0) * CIN, 0, 0);

    __syncthreads();  // previous tap's LDS readers are done
#pragma unroll
    for (int s = 0; s < CIN / 64; ++s) {
      const int h0 = (g + 8 * s) * 8;      // halves offset within the row
      const _Float16* gsrc = in + ibase + h0;
      _Float16* ldst = &sA[p * CIN + h0];
#if HAVE_ASYNC_LDS
      if (inb) async_copy16(gsrc, ldst);
      else     *(v8h*)ldst = (v8h){};
#else
      v8h v = {};
      if (inb) v = *(const v8h*)gsrc;
      *(v8h*)ldst = v;
#endif
    }
#if HAVE_ASYNC_LDS
    __builtin_amdgcn_s_wait_asynccnt(0);
#endif
    __syncthreads();  // tap chunk visible to all waves

#pragma unroll
    for (int cc = 0; cc < CIN / 32; ++cc) {
      v16h a  = load_fragA_k32(sA, CIN, cc * 32);
      v16h bf = load_fragB_k32(wbase, CIN, cc * 32);
      acc = __builtin_amdgcn_wmma_f32_16x16x32_f16(false, a, false, bf,
                                                   (short)0, acc, false, false);
    }
  }
  // epilogue: bias + relu, scatter to NHWC f16
  const int n  = lane & 15;
  const int co = co0 + n;
  const float bv = bias[co];
  const int mh = (lane < 16) ? 0 : 8;
#pragma unroll
  for (int r = 0; r < 8; ++r) {
    const int opix = pix0 + r + mh;
    float v = acc[r] + bv;
    v = v > 0.f ? v : 0.f;
    out[(size_t)opix * COUT + co] = (_Float16)v;
  }
}

// ----- RPN 1x1 heads: N=18/36, K=256 -> latency-bound VALU ------------------
__global__ __launch_bounds__(256)
void rpn_head_kernel(const _Float16* __restrict__ f4, const float* __restrict__ wc,
                     const float* __restrict__ bc, const float* __restrict__ wb,
                     const float* __restrict__ bb, float* __restrict__ cls_out,
                     float* __restrict__ bbox_out) {
  int idx = blockIdx.x * 256 + threadIdx.x;  // 4096 * 54
  if (idx >= 4096 * 54) return;
  int oc = idx % 54;
  int pix = idx / 54;
  const _Float16* fp = f4 + (size_t)pix * 256;
  if (oc < 18) {
    float acc = bc[oc];
    for (int k = 0; k < 256; ++k) acc += (float)fp[k] * wc[k * 18 + oc];
    cls_out[(size_t)pix * 18 + oc] = acc;
  } else {
    int o = oc - 18;
    float acc = bb[o];
    for (int k = 0; k < 256; ++k) acc += (float)fp[k] * wb[k * 36 + o];
    bbox_out[(size_t)pix * 36 + o] = acc;
  }
}

// ----- proposal decode + fg softmax ----------------------------------------
__constant__ float ANCH_H[9] = {64.f, 64.f, 128.f, 128.f, 128.f, 256.f, 256.f, 256.f, 512.f};
__constant__ float ANCH_W[9] = {64.f, 128.f, 64.f, 128.f, 256.f, 128.f, 256.f, 512.f, 256.f};

__global__ __launch_bounds__(256)
void decode_kernel(const float* __restrict__ cls, const float* __restrict__ bbox,
                   float* __restrict__ proposals, float* __restrict__ fgprob) {
  int idx = blockIdx.x * 256 + threadIdx.x;  // 4*1024*9
  if (idx >= 4 * 1024 * 9) return;
  int a = idx % 9;
  int loc = (idx / 9) % 1024;
  int b = idx / (9 * 1024);
  int y = loc / 32, x = loc % 32;
  const float* d = bbox + (size_t)(b * 1024 + loc) * 36 + a * 4;
  float ah = ANCH_H[a], aw = ANCH_W[a];
  float cy = (y + 0.5f) * 16.f, cx = (x + 0.5f) * 16.f;
  float yc = cy + d[0] * ah;
  float xc = cx + d[1] * aw;
  float t2 = fminf(fmaxf(d[2], -4.f), 4.f);
  float t3 = fminf(fmaxf(d[3], -4.f), 4.f);
  float hh = ah * __expf(t2);
  float ww = aw * __expf(t3);
  const float inv = 1.f / 512.f;
  float y1 = fminf(fmaxf(yc - hh * 0.5f, 0.f), 512.f) * inv;
  float x1 = fminf(fmaxf(xc - ww * 0.5f, 0.f), 512.f) * inv;
  float y2 = fminf(fmaxf(yc + hh * 0.5f, 0.f), 512.f) * inv;
  float x2 = fminf(fmaxf(xc + ww * 0.5f, 0.f), 512.f) * inv;
  float* pp = proposals + (size_t)idx * 4;
  pp[0] = y1; pp[1] = x1; pp[2] = y2; pp[3] = x2;
  const float* lg = cls + (size_t)(b * 1024 + loc) * 18 + a * 2;
  float m = fmaxf(lg[0], lg[1]);
  float e0 = __expf(lg[0] - m), e1 = __expf(lg[1] - m);
  fgprob[idx] = e1 / (e0 + e1);
}

// ----- top-k 300 per batch: block-wide iterative argmax ---------------------
__global__ __launch_bounds__(256)
void topk_kernel(const float* __restrict__ fgprob, const float* __restrict__ proposals,
                 float* __restrict__ probtmp, float* __restrict__ out_prop,
                 float* __restrict__ out_prob) {
  const int b = blockIdx.x;
  const int tid = threadIdx.x;
  __shared__ float sv[256];
  __shared__ int si[256];
  const int N = 9216;
  const float* src = fgprob + (size_t)b * N;
  float* tmp = probtmp + (size_t)b * N;
  for (int i = tid; i < N; i += 256) tmp[i] = src[i];
  __syncthreads();
  for (int it = 0; it < 300; ++it) {
    float bv = -1e30f;
    int bi = 0x7fffffff;
    for (int i = tid; i < N; i += 256) {
      float v = tmp[i];
      if (v > bv || (v == bv && i < bi)) { bv = v; bi = i; }
    }
    sv[tid] = bv; si[tid] = bi;
    __syncthreads();
    for (int s = 128; s > 0; s >>= 1) {
      if (tid < s) {
        float ov = sv[tid + s]; int oi = si[tid + s];
        if (ov > sv[tid] || (ov == sv[tid] && oi < si[tid])) { sv[tid] = ov; si[tid] = oi; }
      }
      __syncthreads();
    }
    if (tid == 0) {
      int sel = si[0];
      out_prob[b * 300 + it] = sv[0];
      const float* pp = proposals + ((size_t)b * N + sel) * 4;
      float* op = out_prop + (size_t)(b * 300 + it) * 4;
      op[0] = pp[0]; op[1] = pp[1]; op[2] = pp[2]; op[3] = pp[3];
      tmp[sel] = -1e30f;
    }
    __syncthreads();
  }
}

// ----- ROI max pool (3x3 cells, 2x2 samples) -> f16 [1200][2304] ------------
__global__ __launch_bounds__(256)
void roipool_kernel(const _Float16* __restrict__ f4, const float* __restrict__ rois,
                    _Float16* __restrict__ pooled) {
  const int rk = blockIdx.x;  // b*300 + k
  const int b = rk / 300;
  const int c = threadIdx.x;  // channel
  const float* p = rois + (size_t)rk * 4;
  const float ys0 = p[0] * 32.f, xs0 = p[1] * 32.f;
  const float ys1 = p[2] * 32.f, xs1 = p[3] * 32.f;
  const float bh = (ys1 - ys0) * (1.f / 3.f);
  const float bw = (xs1 - xs0) * (1.f / 3.f);
  const _Float16* fb = f4 + (size_t)b * 32 * 32 * 256 + c;
#pragma unroll
  for (int py = 0; py < 3; ++py) {
#pragma unroll
    for (int px = 0; px < 3; ++px) {
      float m = -1e30f;
#pragma unroll
      for (int sy = 0; sy < 2; ++sy) {
        float ys = ys0 + (py + (sy + 0.5f) * 0.5f) * bh;
        int yi = (int)floorf(ys);
        yi = yi < 0 ? 0 : (yi > 31 ? 31 : yi);
#pragma unroll
        for (int sx = 0; sx < 2; ++sx) {
          float xs = xs0 + (px + (sx + 0.5f) * 0.5f) * bw;
          int xi = (int)floorf(xs);
          xi = xi < 0 ? 0 : (xi > 31 ? 31 : xi);
          float v = (float)fb[((size_t)yi * 32 + xi) * 256];
          m = v > m ? v : m;
        }
      }
      pooled[((size_t)rk * 9 + (py * 3 + px)) * 256 + c] = (_Float16)m;
    }
  }
}

// ----- generic wave-level WMMA GEMM: C[M][N] = relu?(A[M][K] * B^T + bias) --
__global__ __launch_bounds__(32)
void fc_gemm_wmma_kernel(const _Float16* __restrict__ A,   // [M][K] f16
                         const _Float16* __restrict__ Bp,  // [N][K] f16
                         const float* __restrict__ bias,
                         float* __restrict__ C, int M, int N, int K, int relu) {
  const int mt = blockIdx.x, nt = blockIdx.y;
  const int lane = threadIdx.x & 31;
  v8f acc = {};
  const _Float16* Abase = A + (size_t)mt * 16 * K;
  const _Float16* Bbase = Bp + (size_t)nt * 16 * K;
  for (int k0 = 0; k0 < K; k0 += 32) {
    v16h a = load_fragA_k32(Abase, K, k0);
    v16h b = load_fragB_k32(Bbase, K, k0);
    acc = __builtin_amdgcn_wmma_f32_16x16x32_f16(false, a, false, b,
                                                 (short)0, acc, false, false);
  }
  const int n = nt * 16 + (lane & 15);
  const float bv = bias[n];
  const int mh = (lane < 16) ? 0 : 8;
#pragma unroll
  for (int r = 0; r < 8; ++r) {
    int m = mt * 16 + r + mh;
    float v = acc[r] + bv;
    if (relu) v = v > 0.f ? v : 0.f;
    C[(size_t)m * N + n] = v;
  }
}

// ----- detection heads: N=20 (sigmoid) and N=4 ------------------------------
__global__ __launch_bounds__(256)
void final_head_kernel(const float* __restrict__ fc2, const float* __restrict__ wcls,
                       const float* __restrict__ bcls, const float* __restrict__ wbox,
                       const float* __restrict__ bbox, float* __restrict__ cls_out,
                       float* __restrict__ box_out) {
  int idx = blockIdx.x * 256 + threadIdx.x;  // 1200 * 24
  if (idx >= 1200 * 24) return;
  int oc = idx % 24;
  int row = idx / 24;
  const float* f = fc2 + (size_t)row * 256;
  if (oc < 20) {
    float acc = bcls[oc];
    for (int k = 0; k < 256; ++k) acc += f[k] * wcls[k * 20 + oc];
    cls_out[(size_t)row * 20 + oc] = 1.f / (1.f + __expf(-acc));
  } else {
    int o = oc - 20;
    float acc = bbox[o];
    for (int k = 0; k < 256; ++k) acc += f[k] * wbox[k * 4 + o];
    box_out[(size_t)row * 4 + o] = acc;
  }
}

// ---------------------------------------------------------------------------
extern "C" void kernel_launch(void* const* d_in, const int* in_sizes, int n_in,
                              void* d_out, int out_size, void* d_ws, size_t ws_size,
                              hipStream_t stream) {
  (void)in_sizes; (void)n_in; (void)out_size; (void)ws_size;
  const float* x       = (const float*)d_in[0];
  const float* w1      = (const float*)d_in[1];
  const float* b1      = (const float*)d_in[2];
  const float* w2      = (const float*)d_in[3];
  const float* b2      = (const float*)d_in[4];
  const float* w3      = (const float*)d_in[5];
  const float* b3      = (const float*)d_in[6];
  const float* w4      = (const float*)d_in[7];
  const float* b4      = (const float*)d_in[8];
  const float* w_cls   = (const float*)d_in[9];
  const float* b_cls   = (const float*)d_in[10];
  const float* w_bbox  = (const float*)d_in[11];
  const float* b_bbox  = (const float*)d_in[12];
  // d_in[13]/d_in[14] = w_fc1/b_fc1: dead in the reference (fc1 never returned)
  const float* w_fc2   = (const float*)d_in[15];
  const float* b_fc2   = (const float*)d_in[16];
  const float* w_clsfc = (const float*)d_in[17];
  const float* b_clsfc = (const float*)d_in[18];
  const float* w_boxfc = (const float*)d_in[19];
  const float* b_boxfc = (const float*)d_in[20];

  char* ws = (char*)d_ws;
  size_t off = 0;
  auto alloc = [&](size_t bytes) {
    void* p = ws + off;
    off = (off + bytes + 255) & ~(size_t)255;
    return p;
  };

  // regionA holds f1, then f3, then pooled (each dead before the next writer)
  // regionB holds f2, then f4
  const size_t f1_bytes = (size_t)4 * 256 * 256 * 64 * 2;   // 33.5 MB
  const size_t f2_bytes = (size_t)4 * 128 * 128 * 128 * 2;  // 16.8 MB
  char* regionA = (char*)alloc(f1_bytes);
  char* regionB = (char*)alloc(f2_bytes);
  _Float16* f1 = (_Float16*)regionA;
  _Float16* f2 = (_Float16*)regionB;
  _Float16* f3 = (_Float16*)regionA;
  _Float16* f4 = (_Float16*)regionB;
  _Float16* pooled = (_Float16*)regionA;  // 1200*2304 f16 = 5.5 MB

  _Float16* wp2    = (_Float16*)alloc((size_t)9 * 64 * 128 * 2);
  _Float16* wp3    = (_Float16*)alloc((size_t)9 * 128 * 256 * 2);
  _Float16* wp4    = (_Float16*)alloc((size_t)9 * 256 * 256 * 2);
  _Float16* wfc2p  = (_Float16*)alloc((size_t)2304 * 256 * 2);
  float* cls_conv  = (float*)alloc((size_t)4096 * 18 * 4);
  float* bbox_conv = (float*)alloc((size_t)4096 * 36 * 4);
  float* proposals = (float*)alloc((size_t)36864 * 4 * 4);
  float* fgprob    = (float*)alloc((size_t)36864 * 4);
  float* probtmp   = (float*)alloc((size_t)36864 * 4);
  float* fc2out    = (float*)alloc((size_t)1200 * 256 * 4);

  float* out = (float*)d_out;
  float* out_prop = out;           // (4,300,4)  = 4800
  float* out_prob = out + 4800;    // (4,300)    = 1200
  float* out_cls  = out + 6000;    // (4,300,20) = 24000
  float* out_box  = out + 30000;   // (4,300,4)  = 4800

  // weight packing (f32 -> f16, B-fragment friendly layouts)
  pack_conv_w<<<(9 * 64 * 128 + 255) / 256, 256, 0, stream>>>(w2, wp2, 64, 128);
  pack_conv_w<<<(9 * 128 * 256 + 255) / 256, 256, 0, stream>>>(w3, wp3, 128, 256);
  pack_conv_w<<<(9 * 256 * 256 + 255) / 256, 256, 0, stream>>>(w4, wp4, 256, 256);
  pack_fc_w<<<(2304 * 256 + 255) / 256, 256, 0, stream>>>(w_fc2, wfc2p, 2304, 256);

  // backbone
  conv1_direct_kernel<<<(4 * 256 * 256 * 64) / 256, 256, 0, stream>>>(
      x, w1, b1, f1, 4 * 256 * 256 * 64);
  conv3x3_s2_wmma_kernel<64, 128><<<dim3(4096, 2), 128, 0, stream>>>(
      f1, wp2, b2, f2, 256, 256, 128, 128);
  conv3x3_s2_wmma_kernel<128, 256><<<dim3(1024, 4), 128, 0, stream>>>(
      f2, wp3, b3, f3, 128, 128, 64, 64);
  conv3x3_s2_wmma_kernel<256, 256><<<dim3(256, 4), 128, 0, stream>>>(
      f3, wp4, b4, f4, 64, 64, 32, 32);

  // RPN + proposals
  rpn_head_kernel<<<(4096 * 54 + 255) / 256, 256, 0, stream>>>(
      f4, w_cls, b_cls, w_bbox, b_bbox, cls_conv, bbox_conv);
  decode_kernel<<<(36864 + 255) / 256, 256, 0, stream>>>(
      cls_conv, bbox_conv, proposals, fgprob);
  topk_kernel<<<4, 256, 0, stream>>>(fgprob, proposals, probtmp, out_prop, out_prob);

  // ROI pool + detection head (fc1 skipped: dead in reference)
  roipool_kernel<<<1200, 256, 0, stream>>>(f4, out_prop, pooled);
  fc_gemm_wmma_kernel<<<dim3(75, 16), 32, 0, stream>>>(
      pooled, wfc2p, b_fc2, fc2out, 1200, 256, 2304, 1);
  final_head_kernel<<<(1200 * 24 + 255) / 256, 256, 0, stream>>>(
      fc2out, w_clsfc, b_clsfc, w_boxfc, b_boxfc, out_cls, out_box);
}
